// SmolVLMWithExpertModel_51831665328384
// MI455X (gfx1250) — matrix operand order repaired and checked by reading, not compiled, not run on hardware
//
#include <hip/hip_runtime.h>
#include <hip/hip_bf16.h>

// ---------------------------------------------------------------------------
// Model dims (compile-time constants)
// ---------------------------------------------------------------------------
#define LNUM 16
#define DMODEL 960
#define HQ 15
#define HKV 5
#define DH 64
#define FF 2560
#define BB 2
#define SS 1024
#define NTOK (BB * SS)            // 2048
#define QKV_LD (HQ*DH + 2*HKV*DH) // 1600
#define EPSV 1e-5f

typedef __attribute__((ext_vector_type(16))) __bf16 bf16x16;
typedef __attribute__((ext_vector_type(8)))  float  f32x8;

// ---------------------------------------------------------------------------
// WMMA fragment loaders (bf16, 16x16x32, wave32; contiguous k = 16*(lane/16) + s for A and B)
// ---------------------------------------------------------------------------
// A-matrix 16x32 (MxK), element (m,c) at base[m*stride + c]
static __device__ __forceinline__ bf16x16 load_a_rowmajor(const __bf16* base, int stride, int lane) {
    int m  = lane & 15;
    int kh = (lane >> 4) * 16;
    bf16x16 r;
#pragma unroll
    for (int i = 0; i < 8; ++i) {
        int kb = 2 * i + kh;
        r[2 * i]     = base[m * stride + kb];
        r[2 * i + 1] = base[m * stride + kb + 1];
    }
    return r;
}

// A-matrix 16x32 where element (m,c) is stored at base[c*stride + m] (transposed source)
static __device__ __forceinline__ bf16x16 load_a_colmajor(const __bf16* base, int stride, int lane) {
    int m  = lane & 15;
    int kh = (lane >> 4) * 16;
    bf16x16 r;
#pragma unroll
    for (int i = 0; i < 8; ++i) {
        int kb = 2 * i + kh;
        r[2 * i]     = base[kb * stride + m];
        r[2 * i + 1] = base[(kb + 1) * stride + m];
    }
    return r;
}

// B-matrix 32x16 (KxN), element (k,n) stored at base[n*stride + k]  (N-major, k contiguous)
static __device__ __forceinline__ bf16x16 load_b_nmajor(const __bf16* base, int stride, int lane) {
    int n  = lane & 15;
    int kh = (lane >> 4) * 16;
    bf16x16 r;
#pragma unroll
    for (int i = 0; i < 8; ++i) {
        r[2 * i]     = base[n * stride + kh + 2 * i];
        r[2 * i + 1] = base[n * stride + kh + 2 * i + 1];
    }
    return r;
}

// B-matrix 32x16 (KxN), element (k,n) stored at base[k*stride + n]  (K-major)
static __device__ __forceinline__ bf16x16 load_b_kmajor(const __bf16* base, int stride, int lane) {
    int n  = lane & 15;
    int kh = (lane >> 4) * 16;
    bf16x16 r;
#pragma unroll
    for (int i = 0; i < 8; ++i) {
        r[2 * i]     = base[(kh + 2 * i) * stride + n];
        r[2 * i + 1] = base[(kh + 2 * i + 1) * stride + n];
    }
    return r;
}

static __device__ __forceinline__ f32x8 wmma_bf16(bf16x16 a, bf16x16 b, f32x8 c) {
    return __builtin_amdgcn_wmma_f32_16x16x32_bf16(false, a, false, b, (short)0, c, false, false);
}

static __device__ __forceinline__ f32x8 zero8() {
    return (f32x8){0.f,0.f,0.f,0.f,0.f,0.f,0.f,0.f};
}

// ---------------------------------------------------------------------------
// GEMM: C[M,N] = A[M,K] (f32) @ B[K,N] (f32)  (+ residual), bf16 WMMA inside.
// Block: 256 threads (8 waves). Block tile 128x128, K-step 32.
// Wave layout: 4(M) x 2(N); each wave owns a 32x64 strip -> 8 WMMAs / K-step.
// Global loads are float4 (global_load_b128); next tiles prefetched.
// ---------------------------------------------------------------------------
__global__ __launch_bounds__(256) void gemm_bf16_kernel(
        const float* __restrict__ A, const float* __restrict__ Bw,
        const float* __restrict__ residual, float* __restrict__ C,
        int M, int N, int K, int lda, int ldb, int ldc) {
    __shared__ __bf16 aLds[128][34];
    __shared__ __bf16 bLds[128][34];   // stored transposed: [n][k]

    const int tid  = threadIdx.x;
    const int lane = tid & 31;
    const int wave = tid >> 5;
    const int wm   = (wave & 3) * 32;      // 32-row strip within block tile
    const int wn   = (wave >> 2) * 64;     // 64-col strip within block tile
    const int bm   = blockIdx.y * 128;
    const int bn   = blockIdx.x * 128;

    f32x8 acc[2][4];
#pragma unroll
    for (int i = 0; i < 2; ++i)
#pragma unroll
        for (int t = 0; t < 4; ++t) acc[i][t] = zero8();

    for (int k0 = 0; k0 < K; k0 += 32) {
        // Stage A tile 128x32 (f32 -> bf16), float4 global loads
#pragma unroll
        for (int it = 0; it < 4; ++it) {
            int idx = tid + it * 256;               // 1024 float4 = 128*32 floats
            int mL = idx >> 3, kL = (idx & 7) * 4;
            float4 v = *(const float4*)&A[(size_t)(bm + mL) * lda + (k0 + kL)];
            aLds[mL][kL + 0] = (__bf16)v.x;
            aLds[mL][kL + 1] = (__bf16)v.y;
            aLds[mL][kL + 2] = (__bf16)v.z;
            aLds[mL][kL + 3] = (__bf16)v.w;
        }
        // Stage B tile 32x128 transposed (f32 -> bf16), float4 global loads, guard N
#pragma unroll
        for (int it = 0; it < 4; ++it) {
            int idx = tid + it * 256;               // 1024 float4 = 32*128 floats
            int kL = idx >> 5, nL = (idx & 31) * 4;
            int gN = bn + nL;
            float4 v;
            if (gN < N) v = *(const float4*)&Bw[(size_t)(k0 + kL) * ldb + gN];
            else        v = make_float4(0.f, 0.f, 0.f, 0.f);
            bLds[nL + 0][kL] = (__bf16)v.x;
            bLds[nL + 1][kL] = (__bf16)v.y;
            bLds[nL + 2][kL] = (__bf16)v.z;
            bLds[nL + 3][kL] = (__bf16)v.w;
        }
        // Prefetch next K-step tiles into cache (global_prefetch_b8)
        if (k0 + 32 < K) {
            int idx = tid;
            int mL = idx >> 3, kL = (idx & 7) * 4;
            __builtin_prefetch(&A[(size_t)(bm + mL) * lda + (k0 + 32 + kL)], 0, 1);
            int kB = idx >> 5, nB = (idx & 31) * 4;
            if (bn + nB < N)
                __builtin_prefetch(&Bw[(size_t)(k0 + 32 + kB) * ldb + bn + nB], 0, 1);
        }
        __syncthreads();

        bf16x16 a0 = load_a_rowmajor(&aLds[wm][0],      34, lane);
        bf16x16 a1 = load_a_rowmajor(&aLds[wm + 16][0], 34, lane);
#pragma unroll
        for (int t = 0; t < 4; ++t) {
            bf16x16 bfrag = load_b_nmajor(&bLds[wn + t * 16][0], 34, lane);
            acc[0][t] = wmma_bf16(a0, bfrag, acc[0][t]);
            acc[1][t] = wmma_bf16(a1, bfrag, acc[1][t]);
        }
        __syncthreads();
    }

    // Epilogue: C layout — lanes 0-15: N=lane, M=r; lanes 16-31: N=lane-16, M=r+8
#pragma unroll
    for (int i = 0; i < 2; ++i) {
#pragma unroll
        for (int t = 0; t < 4; ++t) {
#pragma unroll
            for (int r = 0; r < 8; ++r) {
                int row = bm + wm + i * 16 + r + 8 * (lane >> 4);
                int col = bn + wn + t * 16 + (lane & 15);
                if (col < N) {
                    float v = acc[i][t][r];
                    if (residual) v += residual[(size_t)row * ldc + col];
                    C[(size_t)row * ldc + col] = v;
                }
            }
        }
    }
}

// ---------------------------------------------------------------------------
// RMSNorm: one block per token row
// ---------------------------------------------------------------------------
__global__ __launch_bounds__(256) void rmsnorm_kernel(
        const float* __restrict__ x, const float* __restrict__ w,
        float* __restrict__ y) {
    __shared__ float red[256];
    const int row = blockIdx.x;
    const float* xr = x + (size_t)row * DMODEL;
    float s = 0.f;
    for (int i = threadIdx.x; i < DMODEL; i += 256) { float v = xr[i]; s += v * v; }
    red[threadIdx.x] = s;
    __syncthreads();
    for (int off = 128; off > 0; off >>= 1) {
        if (threadIdx.x < off) red[threadIdx.x] += red[threadIdx.x + off];
        __syncthreads();
    }
    float rs = rsqrtf(red[0] / (float)DMODEL + EPSV);
    float* yr = y + (size_t)row * DMODEL;
    for (int i = threadIdx.x; i < DMODEL; i += 256) yr[i] = xr[i] * rs * w[i];
}

// ---------------------------------------------------------------------------
// RoPE applied in place to the fused qkv buffer (q heads 0..14, k heads 0..4)
// ---------------------------------------------------------------------------
__global__ __launch_bounds__(256) void rope_kernel(float* __restrict__ qkv,
                                                   const int* __restrict__ pos_ids) {
    const int total = NTOK * (HQ + HKV) * (DH / 2);
    int gid = blockIdx.x * 256 + threadIdx.x;
    if (gid >= total) return;
    int i   = gid & 31;                 // 0..31 (d_half)
    int hh  = (gid >> 5) % (HQ + HKV);  // 0..19
    int row = gid / ((HQ + HKV) * 32);
    int col = (hh < HQ) ? hh * DH : HQ * DH + (hh - HQ) * DH;
    float p  = (float)pos_ids[row];
    float ts = __powf(10000.0f, (2.0f / (float)DH) * (float)i);
    float rad = p / ts;
    float sn, cs;
    __sincosf(rad, &sn, &cs);
    float* b = qkv + (size_t)row * QKV_LD + col;
    float x1 = b[i], x2 = b[i + 32];
    b[i]      = x1 * cs - x2 * sn;
    b[i + 32] = x2 * cs + x1 * sn;
}

// ---------------------------------------------------------------------------
// Flash attention, one wave per (qtile 16 rows, head, batch).
// Scores computed transposed (S^T = K*Q^T) so softmax is per-lane + one
// shfl_xor(16); PV as out^T = V^T * P^T. All matmuls via bf16 WMMA.
// Staging loads are float4 (global_load_b128).
// ---------------------------------------------------------------------------
__global__ __launch_bounds__(32) void attn_kernel(const float* __restrict__ qkv,
                                                  float* __restrict__ ctx) {
    __shared__ __bf16 qLds[16][66];
    __shared__ __bf16 kLds[32][66];
    __shared__ __bf16 vLds[32][66];
    __shared__ __bf16 pLds[32][18];

    const int lane = threadIdx.x;
    const int qt = blockIdx.x, h = blockIdx.y, b = blockIdx.z;
    const int kvh = h / (HQ / HKV);
    const size_t rowBase = (size_t)(b * SS + qt * 16);

    // Stage Q tile (16 x 64), float4 loads
    const float* qp = qkv + rowBase * QKV_LD + h * DH;
#pragma unroll
    for (int it = 0; it < 8; ++it) {
        int idx = lane + it * 32;                   // 256 float4 = 16*64
        int r = idx >> 4, c = (idx & 15) * 4;
        float4 v = *(const float4*)&qp[(size_t)r * QKV_LD + c];
        qLds[r][c + 0] = (__bf16)v.x;
        qLds[r][c + 1] = (__bf16)v.y;
        qLds[r][c + 2] = (__bf16)v.z;
        qLds[r][c + 3] = (__bf16)v.w;
    }

    f32x8 acc[4];
#pragma unroll
    for (int t = 0; t < 4; ++t) acc[t] = zero8();
    float m_i = -1e30f, l_i = 0.f;
    const int qg = qt * 16 + (lane & 15);
    const int nChunks = (qt * 16 + 16 + 31) / 32;

    for (int ch = 0; ch < nChunks; ++ch) {
        const int ks = ch * 32;
        const float* kp = qkv + (size_t)(b * SS + ks) * QKV_LD + HQ * DH + kvh * DH;
        const float* vp = qkv + (size_t)(b * SS + ks) * QKV_LD + (HQ + HKV) * DH + kvh * DH;
#pragma unroll
        for (int it = 0; it < 16; ++it) {           // 512 float4 = 32*64 each
            int idx = lane + it * 32;
            int r = idx >> 4, c = (idx & 15) * 4;
            float4 kv4 = *(const float4*)&kp[(size_t)r * QKV_LD + c];
            float4 vv4 = *(const float4*)&vp[(size_t)r * QKV_LD + c];
            kLds[r][c + 0] = (__bf16)kv4.x;
            kLds[r][c + 1] = (__bf16)kv4.y;
            kLds[r][c + 2] = (__bf16)kv4.z;
            kLds[r][c + 3] = (__bf16)kv4.w;
            vLds[r][c + 0] = (__bf16)vv4.x;
            vLds[r][c + 1] = (__bf16)vv4.y;
            vLds[r][c + 2] = (__bf16)vv4.z;
            vLds[r][c + 3] = (__bf16)vv4.w;
        }
        __syncthreads();

        // S^T = K_chunk (32 x 64) @ Q^T (64 x 16) : two M-tiles, dh in 2 steps
        f32x8 s0 = zero8();
        f32x8 s1 = zero8();
#pragma unroll
        for (int half = 0; half < 2; ++half) {
            bf16x16 bq = load_b_nmajor(&qLds[0][half * 32], 66, lane);
            bf16x16 a0 = load_a_rowmajor(&kLds[0][half * 32], 66, lane);
            bf16x16 a1 = load_a_rowmajor(&kLds[16][half * 32], 66, lane);
            s0 = wmma_bf16(a0, bq, s0);
            s1 = wmma_bf16(a1, bq, s1);
        }

        // scale + causal mask (lane -> q column, VGPR -> k row)
        float cmax = -1e30f;
        float p0[8], p1[8];
#pragma unroll
        for (int r = 0; r < 8; ++r) {
            int kr = r + 8 * (lane >> 4);
            float v0 = (ks + kr      <= qg) ? s0[r] * 0.125f : -1e30f;
            float v1 = (ks + kr + 16 <= qg) ? s1[r] * 0.125f : -1e30f;
            p0[r] = v0; p1[r] = v1;
            cmax = fmaxf(cmax, fmaxf(v0, v1));
        }
        cmax = fmaxf(cmax, __shfl_xor(cmax, 16, 32));
        float m_new = fmaxf(m_i, cmax);
        float alpha = __expf(m_i - m_new);
        float rsum = 0.f;
#pragma unroll
        for (int r = 0; r < 8; ++r) {
            p0[r] = __expf(p0[r] - m_new);
            p1[r] = __expf(p1[r] - m_new);
            rsum += p0[r] + p1[r];
        }
        rsum += __shfl_xor(rsum, 16, 32);
        l_i = l_i * alpha + rsum;
        m_i = m_new;

        // P^T to LDS (32 k-rows x 16 q-cols)
        {
            int q = lane & 15;
#pragma unroll
            for (int r = 0; r < 8; ++r) {
                int kr = r + 8 * (lane >> 4);
                pLds[kr][q]      = (__bf16)p0[r];
                pLds[kr + 16][q] = (__bf16)p1[r];
            }
        }
        __syncthreads();

        // rescale accumulators, then out^T += V^T (dh x 32) @ P^T (32 x 16)
        bf16x16 bp = load_b_kmajor(&pLds[0][0], 18, lane);
#pragma unroll
        for (int t = 0; t < 4; ++t) {
#pragma unroll
            for (int r = 0; r < 8; ++r) acc[t][r] *= alpha;
            bf16x16 av = load_a_colmajor(&vLds[0][t * 16], 66, lane);
            acc[t] = wmma_bf16(av, bp, acc[t]);
        }
        __syncthreads();
    }

    // Store: acc is out^T (dh rows, q cols); ctx[(b,s),(h,dh)]
    float invl = 1.0f / l_i;
    int q = lane & 15;
#pragma unroll
    for (int t = 0; t < 4; ++t) {
#pragma unroll
        for (int r = 0; r < 8; ++r) {
            int dh = t * 16 + r + 8 * (lane >> 4);
            ctx[(rowBase + q) * (size_t)(HQ * DH) + h * DH + dh] = acc[t][r] * invl;
        }
    }
}

// ---------------------------------------------------------------------------
// SiLU(gate) * up, in place into gate buffer
// ---------------------------------------------------------------------------
__global__ __launch_bounds__(256) void silu_mul_kernel(float* __restrict__ g,
                                                       const float* __restrict__ u) {
    size_t i = (size_t)blockIdx.x * 256 + threadIdx.x;
    if (i >= (size_t)NTOK * FF) return;
    float gv = g[i];
    g[i] = (gv / (1.0f + __expf(-gv))) * u[i];
}

// ---------------------------------------------------------------------------
// Host launcher: 16-layer loop
// ---------------------------------------------------------------------------
extern "C" void kernel_launch(void* const* d_in, const int* in_sizes, int n_in,
                              void* d_out, int out_size, void* d_ws, size_t ws_size,
                              hipStream_t stream) {
    const float* prefix = (const float*)d_in[0];
    const float* ln1_w  = (const float*)d_in[1];
    const float* wq     = (const float*)d_in[2];
    const float* wk     = (const float*)d_in[3];
    const float* wv     = (const float*)d_in[4];
    const float* wo     = (const float*)d_in[5];
    const float* ln2_w  = (const float*)d_in[6];
    const float* wg     = (const float*)d_in[7];
    const float* wu     = (const float*)d_in[8];
    const float* wd     = (const float*)d_in[9];
    const int*   pos    = (const int*)d_in[11];
    (void)in_sizes; (void)n_in; (void)out_size; (void)ws_size;

    float* h = (float*)d_out;                       // residual stream, NTOK x D
    float* ws = (float*)d_ws;
    float* hn   = ws;                               // NTOK x D
    float* buf0 = hn   + (size_t)NTOK * DMODEL;     // NTOK x FF (ctx / gate)
    float* buf1 = buf0 + (size_t)NTOK * FF;         // NTOK x FF (up)
    float* qkvb = buf1 + (size_t)NTOK * FF;         // NTOK x 1600

    hipMemcpyAsync(h, prefix, (size_t)NTOK * DMODEL * sizeof(float),
                   hipMemcpyDeviceToDevice, stream);

    const dim3 blk(256);
    for (int l = 0; l < LNUM; ++l) {
        const float* wq_l  = wq + (size_t)l * DMODEL * (HQ * DH);
        const float* wk_l  = wk + (size_t)l * DMODEL * (HKV * DH);
        const float* wv_l  = wv + (size_t)l * DMODEL * (HKV * DH);
        const float* wo_l  = wo + (size_t)l * (HQ * DH) * DMODEL;
        const float* wg_l  = wg + (size_t)l * DMODEL * FF;
        const float* wu_l  = wu + (size_t)l * DMODEL * FF;
        const float* wd_l  = wd + (size_t)l * FF * DMODEL;
        const float* ln1_l = ln1_w + (size_t)l * DMODEL;
        const float* ln2_l = ln2_w + (size_t)l * DMODEL;

        // 1) hn = rmsnorm(h) * ln1
        rmsnorm_kernel<<<NTOK, blk, 0, stream>>>(h, ln1_l, hn);

        // 2) fused qkv buffer: q | k | v  (ldc = 1600)
        gemm_bf16_kernel<<<dim3((HQ*DH + 127) / 128, NTOK / 128), blk, 0, stream>>>(
            hn, wq_l, nullptr, qkvb, NTOK, HQ*DH, DMODEL, DMODEL, HQ*DH, QKV_LD);
        gemm_bf16_kernel<<<dim3((HKV*DH + 127) / 128, NTOK / 128), blk, 0, stream>>>(
            hn, wk_l, nullptr, qkvb + HQ*DH, NTOK, HKV*DH, DMODEL, DMODEL, HKV*DH, QKV_LD);
        gemm_bf16_kernel<<<dim3((HKV*DH + 127) / 128, NTOK / 128), blk, 0, stream>>>(
            hn, wv_l, nullptr, qkvb + HQ*DH + HKV*DH, NTOK, HKV*DH, DMODEL, DMODEL, HKV*DH, QKV_LD);

        // 3) RoPE in place on q & k
        {
            int total = NTOK * (HQ + HKV) * (DH / 2);
            rope_kernel<<<(total + 255) / 256, blk, 0, stream>>>(qkvb, pos);
        }

        // 4) attention -> buf0 (NTOK x 960)
        attn_kernel<<<dim3(SS / 16, HQ, BB), dim3(32), 0, stream>>>(qkvb, buf0);

        // 5) h = h + ctx @ wo
        gemm_bf16_kernel<<<dim3((DMODEL + 127) / 128, NTOK / 128), blk, 0, stream>>>(
            buf0, wo_l, h, h, NTOK, DMODEL, HQ*DH, HQ*DH, DMODEL, DMODEL);

        // 6) hn = rmsnorm(h) * ln2
        rmsnorm_kernel<<<NTOK, blk, 0, stream>>>(h, ln2_l, hn);

        // 7) gate / up
        gemm_bf16_kernel<<<dim3((FF + 127) / 128, NTOK / 128), blk, 0, stream>>>(
            hn, wg_l, nullptr, buf0, NTOK, FF, DMODEL, DMODEL, FF, FF);
        gemm_bf16_kernel<<<dim3((FF + 127) / 128, NTOK / 128), blk, 0, stream>>>(
            hn, wu_l, nullptr, buf1, NTOK, FF, DMODEL, DMODEL, FF, FF);

        // 8) buf0 = silu(gate) * up
        {
            size_t total = (size_t)NTOK * FF;
            silu_mul_kernel<<<(unsigned)((total + 255) / 256), blk, 0, stream>>>(buf0, buf1);
        }

        // 9) h = h + buf0 @ wd
        gemm_bf16_kernel<<<dim3((DMODEL + 127) / 128, NTOK / 128), blk, 0, stream>>>(
            buf0, wd_l, h, h, NTOK, DMODEL, FF, FF, DMODEL, DMODEL);
    }
}
